// Attention_1838246002767
// MI455X (gfx1250) — matrix operand order repaired and checked by reading, not compiled, not run on hardware
//
#include <hip/hip_runtime.h>

typedef __attribute__((ext_vector_type(16))) __bf16 v16bf;
typedef __attribute__((ext_vector_type(8)))  __bf16 v8bf;
typedef __attribute__((ext_vector_type(8)))  float  v8f;
typedef __attribute__((ext_vector_type(4)))  unsigned int vu4;
typedef __attribute__((ext_vector_type(8)))  int vi8;
typedef __attribute__((ext_vector_type(4)))  int vi4;

#define D_MODEL  1024
#define N_HEADS  16
#define HEAD_DIM 64
#define BATCH    2
#define SEQ      2048
#define MROWS    (BATCH * SEQ)   // 4096

#ifndef __has_builtin
#define __has_builtin(x) 0
#endif

__device__ __forceinline__ v8f wmma_bf16(const v16bf& a, const v16bf& b, const v8f& c) {
    return __builtin_amdgcn_wmma_f32_16x16x32_bf16(false, a, false, b, (short)0, c, false, false);
}

// concat two v8bf into one v16bf without element-wise moves
__device__ __forceinline__ v16bf cat16(v8bf lo, v8bf hi) {
    return __builtin_shufflevector(lo, hi, 0,1,2,3,4,5,6,7,8,9,10,11,12,13,14,15);
}

// 16-lane max reduction in the VALU: DPP row_ror butterfly (a DPP row == one C-layout half)
__device__ __forceinline__ float rowmax16_dpp(float x) {
    int t;
    t = __builtin_amdgcn_update_dpp(0, __float_as_int(x), 0x128, 0xf, 0xf, true); // row_ror:8
    x = fmaxf(x, __int_as_float(t));
    t = __builtin_amdgcn_update_dpp(0, __float_as_int(x), 0x124, 0xf, 0xf, true); // row_ror:4
    x = fmaxf(x, __int_as_float(t));
    t = __builtin_amdgcn_update_dpp(0, __float_as_int(x), 0x122, 0xf, 0xf, true); // row_ror:2
    x = fmaxf(x, __int_as_float(t));
    t = __builtin_amdgcn_update_dpp(0, __float_as_int(x), 0x121, 0xf, 0xf, true); // row_ror:1
    x = fmaxf(x, __int_as_float(t));
    return x;
}

// ---------------- GEMM:  C[M,N] = A[M,K] (f32 or bf16) * B[K, col0+N] (f32) + bias ----------------
#define BM 128
#define BN 128
#define BK 32
#define PADA 8
#define PADB 8

template<bool A_BF16, bool OUT_KLAYOUT>
__global__ __launch_bounds__(256)
void gemm_wmma(const void* __restrict__ Aptr,
               const float* __restrict__ B, int ldb, int bcol0,
               const float* __restrict__ bias, int bias0,
               void* __restrict__ Cptr,
               int M, int N, int K)
{
    __shared__ __bf16 As[BM][BK + PADA];   // [m][k]
    __shared__ __bf16 Bs[BN][BK + PADB];   // [n][k] (transposed)

    const int tid  = threadIdx.x;
    const int lane = tid & 31;
    const int wave = tid >> 5;
    const int hlf  = lane >> 4;
    const int l16  = lane & 15;

    const int m0 = blockIdx.y * BM;
    const int n0 = blockIdx.x * BN;
    const int wr = wave >> 1;
    const int wc = wave &  1;

    v8f acc[2][4];
#pragma unroll
    for (int mi = 0; mi < 2; ++mi)
#pragma unroll
        for (int ni = 0; ni < 4; ++ni)
            acc[mi][ni] = (v8f){0.f,0.f,0.f,0.f,0.f,0.f,0.f,0.f};

    for (int k0 = 0; k0 < K; k0 += BK) {
        {   // stage A tile (128x32 -> bf16, row-major)
            const int row = tid >> 1;
            const int cb  = (tid & 1) * 16;
            if (A_BF16) {
                const __bf16* Ab = (const __bf16*)Aptr;
#pragma unroll
                for (int i = 0; i < 2; ++i) {
                    v8bf v = *(const v8bf*)&Ab[(size_t)(m0 + row) * K + k0 + cb + i * 8];
                    *(v8bf*)&As[row][cb + i * 8] = v;
                }
                if (k0 + BK < K)
                    __builtin_prefetch(&Ab[(size_t)(m0 + row) * K + k0 + BK + cb], 0, 0);
            } else {
                const float* Af = (const float*)Aptr;
#pragma unroll
                for (int i = 0; i < 4; ++i) {
                    const float4 v = *(const float4*)&Af[(size_t)(m0 + row) * K + k0 + cb + i * 4];
                    As[row][cb + i * 4 + 0] = (__bf16)v.x;
                    As[row][cb + i * 4 + 1] = (__bf16)v.y;
                    As[row][cb + i * 4 + 2] = (__bf16)v.z;
                    As[row][cb + i * 4 + 3] = (__bf16)v.w;
                }
                if (k0 + BK < K)
                    __builtin_prefetch(&Af[(size_t)(m0 + row) * K + k0 + BK + cb], 0, 0);
            }
        }
        {   // stage B tile (32x128 -> bf16, transposed to [n][k])
            const int row = tid >> 3;
            const int cb  = (tid & 7) * 16;
#pragma unroll
            for (int i = 0; i < 4; ++i) {
                const float4 v = *(const float4*)&B[(size_t)(k0 + row) * ldb + bcol0 + n0 + cb + i * 4];
                Bs[cb + i * 4 + 0][row] = (__bf16)v.x;
                Bs[cb + i * 4 + 1][row] = (__bf16)v.y;
                Bs[cb + i * 4 + 2][row] = (__bf16)v.z;
                Bs[cb + i * 4 + 3][row] = (__bf16)v.w;
            }
            if (k0 + BK < K)
                __builtin_prefetch(&B[(size_t)(k0 + BK + row) * ldb + bcol0 + n0 + cb], 0, 0);
        }
        __syncthreads();

        v16bf afrag[2];
#pragma unroll
        for (int mi = 0; mi < 2; ++mi) {
            const __bf16* ap = &As[wr * 32 + mi * 16 + l16][0];
            afrag[mi] = cat16(*(const v8bf*)&ap[hlf * 8], *(const v8bf*)&ap[16 + hlf * 8]);
        }
        v16bf bfrag[4];
#pragma unroll
        for (int ni = 0; ni < 4; ++ni) {
            const __bf16* bp = &Bs[wc * 64 + ni * 16 + l16][0];
            bfrag[ni] = cat16(*(const v8bf*)&bp[hlf * 8], *(const v8bf*)&bp[16 + hlf * 8]);
        }
#pragma unroll
        for (int mi = 0; mi < 2; ++mi)
#pragma unroll
            for (int ni = 0; ni < 4; ++ni)
                acc[mi][ni] = wmma_bf16(afrag[mi], bfrag[ni], acc[mi][ni]);

        __syncthreads();
    }

#pragma unroll
    for (int mi = 0; mi < 2; ++mi) {
#pragma unroll
        for (int ni = 0; ni < 4; ++ni) {
            const int col = n0 + wc * 64 + ni * 16 + l16;
            const float bv = bias[bias0 + col];
#pragma unroll
            for (int r = 0; r < 8; ++r) {
                const int row = m0 + wr * 32 + mi * 16 + hlf * 8 + r;
                const float val = acc[mi][ni][r] + bv;
                if (OUT_KLAYOUT) {
                    const int bb = row >> 11, ll = row & 2047;
                    const int hh = col >> 6,  dd = col & 63;
                    ((__bf16*)Cptr)[(((size_t)(bb * N_HEADS + hh)) * SEQ + ll) * HEAD_DIM + dd] = (__bf16)val;
                } else {
                    ((float*)Cptr)[(size_t)row * N + col] = val;
                }
            }
        }
    }
}

// ---------------- Flash attention, Q = K = V, per (b,h) ----------------
#define QTILE  128
#define KVTILE 64

__global__ __launch_bounds__(256)
void attn_wmma(const __bf16* __restrict__ Kws, __bf16* __restrict__ Ows)
{
    __shared__ __bf16 KVs[KVTILE][HEAD_DIM + 8];      // [kv][d]  (TDM pad reproduces the +8 stride)
    __shared__ __bf16 KVt[HEAD_DIM + 16][KVTILE + 8]; // [d][kv]; rows 64..79 = ones-tile
    __shared__ __bf16 Ps[8][16][KVTILE];              // per-wave P bounce

    const int tid  = threadIdx.x;
    const int lane = tid & 31;
    const int wave = tid >> 5;
    const int hlf  = lane >> 4;
    const int l16  = lane & 15;

    const int bh = blockIdx.y;
    const int bb = bh >> 4, hh = bh & 15;
    const __bf16* Kh = Kws + (size_t)bh * SEQ * HEAD_DIM;
    const int qbase = blockIdx.x * QTILE + wave * 16;

    {   // ones-tile init: KVt row 64 = 1.0, rows 65..79 = 0
        const int r64 = 64 + (tid >> 4);
        const int c0  = (tid & 15) * 4;
#pragma unroll
        for (int j = 0; j < 4; ++j)
            KVt[r64][c0 + j] = (r64 == 64) ? (__bf16)1.0f : (__bf16)0.0f;
    }

    v16bf qf[2];
#pragma unroll
    for (int t = 0; t < 2; ++t) {
        const __bf16* qp = &Kh[(size_t)(qbase + l16) * HEAD_DIM + t * 32];
        qf[t] = cat16(*(const v8bf*)&qp[hlf * 8], *(const v8bf*)&qp[16 + hlf * 8]);
    }

    float mrun[8];
    v8f   oacc[5];                   // [0..3] = O tiles, [4] = ones-column (row sums)
#pragma unroll
    for (int r = 0; r < 8; ++r) mrun[r] = -1e30f;
#pragma unroll
    for (int ni = 0; ni < 5; ++ni)
        oacc[ni] = (v8f){0.f,0.f,0.f,0.f,0.f,0.f,0.f,0.f};

    const float scale = 0.125f;

#if __has_builtin(__builtin_amdgcn_tensor_load_to_lds)
    const unsigned int lds_kvs = (unsigned int)(size_t)&KVs[0][0];
#endif

    for (int kv = 0; kv < SEQ; kv += KVTILE) {
        __syncthreads();  // previous-iteration readers of KVs/KVt done

#if __has_builtin(__builtin_amdgcn_tensor_load_to_lds)
        // ---- Tensor Data Mover: 64x64 bf16 tile, global -> LDS, +16B pad per 128B row ----
        if (wave == 0) {
            const unsigned long long ga =
                (unsigned long long)(size_t)(Kh + (size_t)kv * HEAD_DIM);
            vu4 g0;
            g0[0] = 1u;                                    // count=1, user mode, no gather
            g0[1] = lds_kvs;                               // lds_addr (bytes)
            g0[2] = (unsigned int)ga;                      // global_addr[31:0]
            g0[3] = (unsigned int)(ga >> 32) | (2u << 30); // global_addr[56:32] | type=2
            vi8 g1;
            g1[0] = (1 << 16)      // data_size = 2 bytes
                  | (1 << 20)      // pad_enable
                  | (4 << 22)      // pad_interval: every 32 DWORDs (=128B row)
                  | (3 << 25);     // pad_amount: 4 DWORDs (=16B -> 72-elem stride)
            g1[1] = (HEAD_DIM << 16);   // tensor_dim0 = 64
            g1[2] = (KVTILE   << 16);   // tensor_dim1 = 64
            g1[3] = (HEAD_DIM << 16);   // tile_dim0 = 64
            g1[4] = KVTILE;             // tile_dim1 = 64, tile_dim2 = 0
            g1[5] = HEAD_DIM;           // tensor_dim0_stride = 64 elements
            g1[6] = 0;
            g1[7] = 0;
            vi4 gz = (vi4){0, 0, 0, 0};
#if defined(__clang_major__) && (__clang_major__ >= 23)
            __builtin_amdgcn_tensor_load_to_lds(g0, g1, gz, gz,
                                                (vi8){0,0,0,0,0,0,0,0}, 0);
#else
            __builtin_amdgcn_tensor_load_to_lds(g0, g1, gz, gz, 0);
#endif
#if __has_builtin(__builtin_amdgcn_s_wait_tensorcnt)
            __builtin_amdgcn_s_wait_tensorcnt(0);
#else
            asm volatile("s_wait_tensorcnt 0x0" ::: "memory");
#endif
        }
        __syncthreads();     // publish TDM-written KVs to all waves
        {   // build transposed copy from LDS
            const int row = tid >> 2;            // kv-local 0..63
            const int cb  = (tid & 3) * 16;
#pragma unroll
            for (int i = 0; i < 2; ++i) {
                const int col = cb + i * 8;
                v8bf v = *(const v8bf*)&KVs[row][col];
#pragma unroll
                for (int j = 0; j < 8; ++j)
                    KVt[col + j][row] = v[j];
            }
        }
#else
        {   // fallback: vector global loads + manual row-major and transposed stores
            const int row = tid >> 2;
            const int cb  = (tid & 3) * 16;
#pragma unroll
            for (int i = 0; i < 2; ++i) {
                const int col = cb + i * 8;
                v8bf v = *(const v8bf*)&Kh[(size_t)(kv + row) * HEAD_DIM + col];
                *(v8bf*)&KVs[row][col] = v;
#pragma unroll
                for (int j = 0; j < 8; ++j)
                    KVt[col + j][row] = v[j];
            }
            if (kv + KVTILE < SEQ)
                __builtin_prefetch(&Kh[(size_t)(kv + KVTILE + row) * HEAD_DIM + cb], 0, 0);
        }
#endif
        __syncthreads();

        // ---- S = Q K^T : four 16x16 tiles ----
        v8f s[4];
#pragma unroll
        for (int nb = 0; nb < 4; ++nb) {
            v8f z = (v8f){0.f,0.f,0.f,0.f,0.f,0.f,0.f,0.f};
#pragma unroll
            for (int t = 0; t < 2; ++t) {
                const __bf16* kp = &KVs[nb * 16 + l16][t * 32];
                v16bf bk = cat16(*(const v8bf*)&kp[hlf * 8], *(const v8bf*)&kp[16 + hlf * 8]);
                z = wmma_bf16(qf[t], bk, z);
            }
            s[nb] = z;
        }

        // ---- online softmax: running max only; sums ride in the ones-column ----
        float pscale[8];
#pragma unroll
        for (int r = 0; r < 8; ++r) {
            const float s0 = s[0][r] * scale;
            const float s1 = s[1][r] * scale;
            const float s2 = s[2][r] * scale;
            const float s3 = s[3][r] * scale;
            float mx = fmaxf(fmaxf(s0, s1), fmaxf(s2, s3));
            mx = rowmax16_dpp(mx);
            const float mnew = fmaxf(mrun[r], mx);
            pscale[r] = __expf(mrun[r] - mnew);
            mrun[r]   = mnew;
            const int prow = hlf * 8 + r;
            Ps[wave][prow][ 0 + l16] = (__bf16)__expf(s0 - mnew);
            Ps[wave][prow][16 + l16] = (__bf16)__expf(s1 - mnew);
            Ps[wave][prow][32 + l16] = (__bf16)__expf(s2 - mnew);
            Ps[wave][prow][48 + l16] = (__bf16)__expf(s3 - mnew);
        }
#pragma unroll
        for (int ni = 0; ni < 5; ++ni)
#pragma unroll
            for (int r = 0; r < 8; ++r)
                oacc[ni][r] *= pscale[r];

        __syncthreads();  // make cross-lane P stores visible

        // ---- O += P [V | 1] ----
#pragma unroll
        for (int t = 0; t < 2; ++t) {
            const __bf16* pp = &Ps[wave][l16][t * 32];
            v16bf pf = cat16(*(const v8bf*)&pp[hlf * 8], *(const v8bf*)&pp[16 + hlf * 8]);
#pragma unroll
            for (int ni = 0; ni < 5; ++ni) {
                const __bf16* vp = &KVt[ni * 16 + l16][t * 32];
                v16bf vf = cat16(*(const v8bf*)&vp[hlf * 8], *(const v8bf*)&vp[16 + hlf * 8]);
                oacc[ni] = wmma_bf16(pf, vf, oacc[ni]);
            }
        }
    }

    const size_t obase = ((size_t)bb * SEQ) * D_MODEL + (size_t)hh * HEAD_DIM;
#pragma unroll
    for (int r = 0; r < 8; ++r) {
        const float denom = __shfl(oacc[4][r], hlf * 16, 32);
        const float inv   = 1.0f / denom;
        const int   q     = qbase + hlf * 8 + r;
#pragma unroll
        for (int ni = 0; ni < 4; ++ni) {
            const int d = ni * 16 + l16;
            Ows[obase + (size_t)q * D_MODEL + d] = (__bf16)(oacc[ni][r] * inv);
        }
    }
}

extern "C" void kernel_launch(void* const* d_in, const int* in_sizes, int n_in,
                              void* d_out, int out_size, void* d_ws, size_t ws_size,
                              hipStream_t stream)
{
    const float* x      = (const float*)d_in[0];
    const float* W_attn = (const float*)d_in[1];
    const float* b_attn = (const float*)d_in[2];
    const float* W_proj = (const float*)d_in[3];
    const float* b_proj = (const float*)d_in[4];
    float* out = (float*)d_out;

    __bf16* Kws = (__bf16*)d_ws;                          // [B][H][L][hd] bf16, 8 MB
    __bf16* Ows = Kws + (size_t)MROWS * D_MODEL;          // [B*L][D]      bf16, 8 MB

    dim3 blk(256);

    dim3 g1(D_MODEL / BN, MROWS / BM);
    gemm_wmma<false, true><<<g1, blk, 0, stream>>>(
        x, W_attn, 3 * D_MODEL, D_MODEL, b_attn, D_MODEL,
        (void*)Kws, MROWS, D_MODEL, D_MODEL);

    dim3 g2(SEQ / QTILE, BATCH * N_HEADS);
    attn_wmma<<<g2, blk, 0, stream>>>(Kws, Ows);

    dim3 g3(D_MODEL / BN, MROWS / BM);
    gemm_wmma<true, false><<<g3, blk, 0, stream>>>(
        (const void*)Ows, W_proj, D_MODEL, 0, b_proj, 0,
        (void*)out, MROWS, D_MODEL, D_MODEL);
}
